// GCN_set2set_62423054680392
// MI455X (gfx1250) — matrix operand order repaired and checked by reading, not compiled, not run on hardware
//
#include <hip/hip_runtime.h>
#include <hip/hip_bf16.h>
#include <stdint.h>

#define NNODES 50000
#define NEDGES 800000
#define NB     256
#define NH     128
#define NLSTM  4
#define PSTEPS 4

typedef __attribute__((ext_vector_type(16))) __bf16 v16bf;
typedef __attribute__((ext_vector_type(8)))  float  v8f;

union BFFrag { uint4 u[2]; v16bf v; };

__device__ __forceinline__ unsigned short f2bf(float f) {
  unsigned u = __float_as_uint(f);
  u = (u + 0x7FFFu + ((u >> 16) & 1u)) >> 16;
  return (unsigned short)u;
}
__device__ __forceinline__ int enc_f(float f) {
  int i = __float_as_int(f);
  return (i >= 0) ? i : (i ^ 0x7FFFFFFF);
}
__device__ __forceinline__ float dec_f(int k) {
  return __int_as_float((k >= 0) ? k : (k ^ 0x7FFFFFFF));
}
__device__ __forceinline__ float sigmf(float x) { return 1.0f / (1.0f + __expf(-x)); }

// ---------------- elementwise / utility kernels ----------------

__global__ void k_fill_f32(float* p, float v, int n) {
  int t = blockIdx.x * blockDim.x + threadIdx.x;
  if (t < n) p[t] = v;
}
__global__ void k_fill_i32(int* p, int v, int n) {
  int t = blockIdx.x * blockDim.x + threadIdx.x;
  if (t < n) p[t] = v;
}
__global__ void k_fill_u16(unsigned short* p, unsigned short v, int n) {
  int t = blockIdx.x * blockDim.x + threadIdx.x;
  if (t < n) p[t] = v;
}
__global__ void k_copy4(const float4* __restrict__ in, float4* __restrict__ out, int n4) {
  int t = blockIdx.x * blockDim.x + threadIdx.x;
  if (t < n4) out[t] = in[t];
}
__global__ void k_cvt_bf16(const float* __restrict__ in, unsigned short* __restrict__ out, int n) {
  int t = blockIdx.x * blockDim.x + threadIdx.x;
  if (t < n) out[t] = f2bf(in[t]);
}

// ---------------- edge aggregation: agg[dst] += x[src] ----------------
// one thread = (edge, float4 quad); agg pre-initialized to x.
__global__ void k_edge_agg(const float* __restrict__ x, const int* __restrict__ src,
                           const int* __restrict__ dst, float* __restrict__ agg) {
  int t = blockIdx.x * blockDim.x + threadIdx.x;
  int e = t >> 5, q = t & 31;
  if (e >= NEDGES) return;
  int sN = src[e], dN = dst[e];
  float4 v = *(const float4*)(x + (size_t)sN * NH + q * 4);
  float* p = agg + (size_t)dN * NH + q * 4;
  atomicAdd(p + 0, v.x); atomicAdd(p + 1, v.y);
  atomicAdd(p + 2, v.z); atomicAdd(p + 3, v.w);
}

// ---------------- WMMA bf16 GEMM: C[M,O] = A[M,K] * W[O,K]^T (+bias | +C) ----
// block = (32, CT); wave tile = 16x16; K,O,M multiples of 32/16/16.
__global__ void k_gemm(const unsigned short* __restrict__ A,
                       const unsigned short* __restrict__ W,
                       const float* __restrict__ bias,
                       float* __restrict__ C,
                       int M, int K, int O, int accum) {
  int lane = threadIdx.x;            // 0..31
  int n    = lane & 15;
  int hi   = lane >> 4;
  int rt   = blockIdx.x;                              // row tile
  int ct   = blockIdx.y * blockDim.y + threadIdx.y;   // col tile
  int o0   = ct * 16;
  const unsigned short* arow = A + (size_t)(rt * 16 + n) * K; // A row m = lane&15
  const unsigned short* wrow = W + (size_t)(o0 + n) * K;      // B col n = lane&15

  v8f acc;
  if (accum) {
#pragma unroll
    for (int d = 0; d < 8; ++d)
      acc[d] = C[(size_t)(rt * 16 + 8 * hi + d) * O + o0 + n];
  } else {
    float bv = bias ? bias[o0 + n] : 0.0f;
#pragma unroll
    for (int d = 0; d < 8; ++d) acc[d] = bv;
  }

  for (int kc = 0; kc < K; kc += 32) {
    BFFrag a, b;
    // A 16x32 bf16 layout: lanes0-15 K∈{0..7,16..23}, lanes16-31 K∈{8..15,24..31}
    a.u[0] = *(const uint4*)(arow + kc + 8 * hi);
    a.u[1] = *(const uint4*)(arow + kc + 16 + 8 * hi);
    // B 32x16 bf16 layout: lanes0-15 K∈{0..7,16..23}, lanes16-31 K∈{8..15,24..31}
    b.u[0] = *(const uint4*)(wrow + kc + 8 * hi);
    b.u[1] = *(const uint4*)(wrow + kc + 16 + 8 * hi);
    acc = __builtin_amdgcn_wmma_f32_16x16x32_bf16(false, a.v, false, b.v,
                                                  (short)0, acc, false, false);
  }
#pragma unroll
  for (int d = 0; d < 8; ++d)
    C[(size_t)(rt * 16 + 8 * hi + d) * O + o0 + n] = acc[d];
}

// ---------------- BatchNorm: column sums / sumsq ----------------
__global__ void k_col_stats(const float* __restrict__ h, float* __restrict__ stats, int M) {
  __shared__ float shs[256];
  __shared__ float shq[256];
  int col = threadIdx.x & 127, half = threadIdx.x >> 7;
  int row0 = blockIdx.x * 64;
  float s = 0.f, s2 = 0.f;
  for (int i = half; i < 64; i += 2) {
    int r = row0 + i;
    if (r < M) { float v = h[(size_t)r * NH + col]; s += v; s2 += v * v; }
  }
  shs[threadIdx.x] = s; shq[threadIdx.x] = s2;
  __syncthreads();
  if (half == 0) {
    atomicAdd(&stats[col],      shs[col] + shs[128 + col]);
    atomicAdd(&stats[128 + col], shq[col] + shq[128 + col]);
  }
}

__global__ void k_bn_finalize(const float* __restrict__ stats,
                              const float* __restrict__ gamma, const float* __restrict__ beta,
                              float* __restrict__ scsh, int M) {
  int c = threadIdx.x;
  if (c >= NH) return;
  float inv = 1.0f / (float)M;
  float mean = stats[c] * inv;
  float var  = stats[128 + c] * inv - mean * mean;
  float scale = gamma[c] * rsqrtf(var + 1e-5f);
  scsh[c] = scale;
  scsh[128 + c] = beta[c] - mean * scale;
}

__global__ void k_bn_relu(const float* __restrict__ h, const float* __restrict__ scsh,
                          float* __restrict__ outf, unsigned short* __restrict__ outbf, int n) {
  int t = blockIdx.x * blockDim.x + threadIdx.x;
  if (t >= n) return;
  int c = t & 127;
  float v = fmaxf(h[t] * scsh[c] + scsh[128 + c], 0.0f);
  outf[t] = v;
  outbf[t] = f2bf(v);
}

// ---------------- LSTM cell elementwise (gate order i,f,g,o) ----------------
__global__ void k_lstm_cell(const float* __restrict__ gates, float* __restrict__ c_state,
                            float* __restrict__ h_state, unsigned short* __restrict__ h_bf) {
  int t = blockIdx.x * blockDim.x + threadIdx.x;
  if (t >= NB * NH) return;
  int b = t >> 7, c = t & 127;
  const float* g = gates + (size_t)b * 512;
  float ig = sigmf(g[c]);
  float fg = sigmf(g[128 + c]);
  float gg = tanhf(g[256 + c]);
  float og = sigmf(g[384 + c]);
  float cn = fg * c_state[t] + ig * gg;
  float hn = og * tanhf(cn);
  c_state[t] = cn;
  h_state[t] = hn;
  h_bf[t] = f2bf(hn);
}

// ---------------- attention: e[n]=dot(x[n],q[batch[n]]) + segment max ------
__global__ void k_att_dot(const float* __restrict__ x, const float* __restrict__ q,
                          const int* __restrict__ batch, float* __restrict__ e,
                          int* __restrict__ m_enc) {
  int t = blockIdx.x * blockDim.x + threadIdx.x;
  int node = t >> 5, lane = threadIdx.x & 31;
  if (node >= NNODES) return;
  int g = batch[node];
  float4 xv = *(const float4*)(x + (size_t)node * NH + lane * 4);
  float4 qv = *(const float4*)(q + (size_t)g * NH + lane * 4);
  float p = xv.x * qv.x + xv.y * qv.y + xv.z * qv.z + xv.w * qv.w;
  for (int off = 16; off; off >>= 1) p += __shfl_xor(p, off, 32);
  if (lane == 0) {
    e[node] = p;
    atomicMax(m_enc + g, enc_f(p));
  }
}

__global__ void k_exp_sum(const float* __restrict__ e, const int* __restrict__ m_enc,
                          const int* __restrict__ batch, float* __restrict__ ee,
                          float* __restrict__ s) {
  int t = blockIdx.x * blockDim.x + threadIdx.x;
  if (t >= NNODES) return;
  int g = batch[t];
  float v = __expf(e[t] - dec_f(m_enc[g]));
  ee[t] = v;
  atomicAdd(&s[g], v);
}

__global__ void k_weighted(const float* __restrict__ x, const float* __restrict__ ee,
                           const float* __restrict__ s, const int* __restrict__ batch,
                           float* __restrict__ r) {
  int t = blockIdx.x * blockDim.x + threadIdx.x;
  int node = t >> 5, q = t & 31;
  if (node >= NNODES) return;
  int g = batch[node];
  float a = ee[node] / s[g];
  float4 v = *(const float4*)(x + (size_t)node * NH + q * 4);
  float* p = r + (size_t)g * NH + q * 4;
  atomicAdd(p + 0, a * v.x); atomicAdd(p + 1, a * v.y);
  atomicAdd(p + 2, a * v.z); atomicAdd(p + 3, a * v.w);
}

__global__ void k_build_qstar(const float* __restrict__ q, const float* __restrict__ r,
                              float* __restrict__ qstar) {
  int t = blockIdx.x * blockDim.x + threadIdx.x;
  if (t >= NB * 2 * NH) return;
  int b = t >> 8, c = t & 255;
  qstar[t] = (c < NH) ? q[b * NH + c] : r[b * NH + (c - NH)];
}

__global__ void k_final(const float* __restrict__ qstar, const float* __restrict__ linW,
                        const float* __restrict__ linb, float* __restrict__ out) {
  int t = blockIdx.x * blockDim.x + threadIdx.x;
  if (t >= NB * 2) return;
  int b = t >> 1, o = t & 1;
  float acc = linb[o];
  const float* qa = qstar + (size_t)b * 256;
  const float* wr = linW + (size_t)o * 256;
  for (int k = 0; k < 256; ++k) acc += qa[k] * wr[k];
  out[b * 2 + o] = acc;
}

// ---------------- host orchestration ----------------

extern "C" void kernel_launch(void* const* d_in, const int* in_sizes, int n_in,
                              void* d_out, int out_size, void* d_ws, size_t ws_size,
                              hipStream_t stream) {
  (void)in_sizes; (void)n_in; (void)out_size; (void)ws_size;

  const float* x      = (const float*)d_in[0];
  const int*   ei     = (const int*)d_in[1];
  const int*   batch  = (const int*)d_in[2];
  const float* gW1[2] = { (const float*)d_in[3],  (const float*)d_in[11] };
  const float* gb1[2] = { (const float*)d_in[4],  (const float*)d_in[12] };
  const float* gg1[2] = { (const float*)d_in[5],  (const float*)d_in[13] };
  const float* gB1[2] = { (const float*)d_in[6],  (const float*)d_in[14] };
  const float* gW2[2] = { (const float*)d_in[7],  (const float*)d_in[15] };
  const float* gb2[2] = { (const float*)d_in[8],  (const float*)d_in[16] };
  const float* gg2[2] = { (const float*)d_in[9],  (const float*)d_in[17] };
  const float* gB2[2] = { (const float*)d_in[10], (const float*)d_in[18] };
  const float* Wih0 = (const float*)d_in[19];
  const float* Whh0 = (const float*)d_in[20];
  const float* b0   = (const float*)d_in[21];
  const float* WihR = (const float*)d_in[22];
  const float* WhhR = (const float*)d_in[23];
  const float* bR   = (const float*)d_in[24];
  const float* linW = (const float*)d_in[25];
  const float* linb = (const float*)d_in[26];
  float* out = (float*)d_out;
  const int* src = ei;
  const int* dst = ei + NEDGES;

  // ---- workspace carve-up ----
  char* base = (char*)d_ws;
  size_t off = 0;
  auto alloc = [&](size_t bytes) -> void* {
    void* p = base + off;
    off = (off + bytes + 255) & ~(size_t)255;
    return p;
  };
  const size_t NHf = (size_t)NNODES * NH;
  float*          work   = (float*)alloc(NHf * 4);
  float*          act    = (float*)alloc(NHf * 4);
  unsigned short* xbf    = (unsigned short*)alloc(NHf * 2);
  unsigned short* wg[4];
  for (int i = 0; i < 4; ++i) wg[i] = (unsigned short*)alloc((size_t)NH * NH * 2);
  unsigned short* wih0bf = (unsigned short*)alloc((size_t)512 * 256 * 2);
  unsigned short* whh0bf = (unsigned short*)alloc((size_t)512 * 128 * 2);
  unsigned short* wihRbf = (unsigned short*)alloc((size_t)3 * 512 * 128 * 2);
  unsigned short* whhRbf = (unsigned short*)alloc((size_t)3 * 512 * 128 * 2);
  float*          stats  = (float*)alloc(256 * 4);
  float*          scsh   = (float*)alloc(256 * 4);
  float*          hs     = (float*)alloc((size_t)NLSTM * NB * NH * 4);
  float*          cs     = (float*)alloc((size_t)NLSTM * NB * NH * 4);
  unsigned short* hbf    = (unsigned short*)alloc((size_t)NLSTM * NB * NH * 2);
  float*          qstar  = (float*)alloc((size_t)NB * 2 * NH * 4);
  unsigned short* qsbf   = (unsigned short*)alloc((size_t)NB * 2 * NH * 2);
  float*          gates  = (float*)alloc((size_t)NB * 512 * 4);
  float*          ebuf   = (float*)alloc((size_t)NNODES * 4);
  float*          eebuf  = (float*)alloc((size_t)NNODES * 4);
  int*            m_enc  = (int*)alloc(NB * 4);
  float*          sbuf   = (float*)alloc(NB * 4);
  float*          rbuf   = (float*)alloc((size_t)NB * NH * 4);

  auto gl = [&](int n) { return dim3((unsigned)((n + 255) / 256)); };
  dim3 B256(256);

  auto gemm = [&](const unsigned short* A, const unsigned short* W, const float* bias,
                  float* C, int M, int K, int O, int accum) {
    dim3 blk(32, 8), grd((unsigned)(M / 16), (unsigned)(O / (16 * 8)));
    k_gemm<<<grd, blk, 0, stream>>>(A, W, bias, C, M, K, O, accum);
  };
  auto cvt = [&](const float* in, unsigned short* o, int n) {
    k_cvt_bf16<<<gl(n), B256, 0, stream>>>(in, o, n);
  };

  // ---- weight conversions ----
  cvt(gW1[0], wg[0], NH * NH); cvt(gW2[0], wg[1], NH * NH);
  cvt(gW1[1], wg[2], NH * NH); cvt(gW2[1], wg[3], NH * NH);
  cvt(Wih0, wih0bf, 512 * 256);
  cvt(Whh0, whh0bf, 512 * 128);
  cvt(WihR, wihRbf, 3 * 512 * 128);
  cvt(WhhR, whhRbf, 3 * 512 * 128);

  // ---- two GIN layers ----
  const float* xin = x;
  for (int layer = 0; layer < 2; ++layer) {
    k_copy4<<<gl((int)(NHf / 4)), B256, 0, stream>>>((const float4*)xin, (float4*)work, (int)(NHf / 4));
    k_edge_agg<<<gl(NEDGES * 32), B256, 0, stream>>>(xin, src, dst, work);
    cvt(work, xbf, (int)NHf);
    gemm(xbf, wg[layer * 2 + 0], gb1[layer], work, NNODES, NH, NH, 0);
    k_fill_f32<<<1, 256, 0, stream>>>(stats, 0.0f, 256);
    k_col_stats<<<gl(NNODES * 4), B256, 0, stream>>>(work, stats, NNODES); // grid=(N+63)/64 blocks => N*4/256
    k_bn_finalize<<<1, 128, 0, stream>>>(stats, gg1[layer], gB1[layer], scsh, NNODES);
    k_bn_relu<<<gl((int)NHf), B256, 0, stream>>>(work, scsh, act, xbf, (int)NHf);
    gemm(xbf, wg[layer * 2 + 1], gb2[layer], work, NNODES, NH, NH, 0);
    k_fill_f32<<<1, 256, 0, stream>>>(stats, 0.0f, 256);
    k_col_stats<<<gl(NNODES * 4), B256, 0, stream>>>(work, stats, NNODES);
    k_bn_finalize<<<1, 128, 0, stream>>>(stats, gg2[layer], gB2[layer], scsh, NNODES);
    k_bn_relu<<<gl((int)NHf), B256, 0, stream>>>(work, scsh, act, xbf, (int)NHf);
    xin = act;
  }

  // ---- Set2Set ----
  k_fill_f32<<<gl(NLSTM * NB * NH), B256, 0, stream>>>(hs, 0.0f, NLSTM * NB * NH);
  k_fill_f32<<<gl(NLSTM * NB * NH), B256, 0, stream>>>(cs, 0.0f, NLSTM * NB * NH);
  k_fill_u16<<<gl(NLSTM * NB * NH), B256, 0, stream>>>(hbf, (unsigned short)0, NLSTM * NB * NH);
  k_fill_f32<<<gl(NB * 2 * NH), B256, 0, stream>>>(qstar, 0.0f, NB * 2 * NH);

  for (int step = 0; step < PSTEPS; ++step) {
    cvt(qstar, qsbf, NB * 2 * NH);
    for (int l = 0; l < NLSTM; ++l) {
      const unsigned short* inp = (l == 0) ? qsbf : (hbf + (size_t)(l - 1) * NB * NH);
      int Kin = (l == 0) ? 2 * NH : NH;
      const unsigned short* wih = (l == 0) ? wih0bf : (wihRbf + (size_t)(l - 1) * 512 * 128);
      const unsigned short* whh = (l == 0) ? whh0bf : (whhRbf + (size_t)(l - 1) * 512 * 128);
      const float* bias = (l == 0) ? b0 : (bR + (size_t)(l - 1) * 512);
      gemm(inp, wih, bias, gates, NB, Kin, 512, 0);
      gemm(hbf + (size_t)l * NB * NH, whh, nullptr, gates, NB, NH, 512, 1);
      k_lstm_cell<<<gl(NB * NH), B256, 0, stream>>>(gates, cs + (size_t)l * NB * NH,
                                                    hs + (size_t)l * NB * NH,
                                                    hbf + (size_t)l * NB * NH);
    }
    const float* q = hs + (size_t)(NLSTM - 1) * NB * NH;
    k_fill_i32<<<1, 256, 0, stream>>>(m_enc, (int)0x807FFFFF, NB); // enc(-inf)
    k_fill_f32<<<1, 256, 0, stream>>>(sbuf, 0.0f, NB);
    k_fill_f32<<<gl(NB * NH), B256, 0, stream>>>(rbuf, 0.0f, NB * NH);
    k_att_dot<<<gl(NNODES * 32), B256, 0, stream>>>(act, q, batch, ebuf, m_enc);
    k_exp_sum<<<gl(NNODES), B256, 0, stream>>>(ebuf, m_enc, batch, eebuf, sbuf);
    k_weighted<<<gl(NNODES * 32), B256, 0, stream>>>(act, eebuf, sbuf, batch, rbuf);
    k_build_qstar<<<gl(NB * 2 * NH), B256, 0, stream>>>(q, rbuf, qstar);
  }

  k_final<<<gl(NB * 2), B256, 0, stream>>>(qstar, linW, linb, out);
}
